// MaskSelfAttentionAggregator_41807211659401
// MI455X (gfx1250) — matrix operand order repaired and checked by reading, not compiled, not run on hardware
//
#include <hip/hip_runtime.h>

// MaskSelfAttentionAggregator for MI455X (gfx1250, wave32, WMMA).
// Pipeline: counting-sort by group (deterministic) -> f16 WMMA projections into
// sorted order -> per-group softmax attention with WMMA scores + WMMA agg.

#define N_PTS   8192
#define DIM     512
#define ADIM    64
#define NGROUPS 64
#define NGCAP   448                      // max supported group size (mean 128, sd ~11 -> 28 sigma margin)
#define NRMAX   (N_PTS + NGROUPS * 32)   // 10240 padded sorted rows (each group padded to mult of 32)

typedef _Float16 f16;
typedef __attribute__((ext_vector_type(16))) _Float16 v16h;
typedef __attribute__((ext_vector_type(8)))  _Float16 v8h;
typedef __attribute__((ext_vector_type(4)))  _Float16 v4h;
typedef __attribute__((ext_vector_type(8)))  float    v8f;

__device__ __forceinline__ v8f wmma_f16(v16h a, v16h b, v8f c) {
  // D = A(16x32 f16) * B(32x16 f16) + C(16x16 f32)
  return __builtin_amdgcn_wmma_f32_16x16x32_f16(false, a, false, b, (short)0, c, false, false);
}

// A-fragment (16x32 f16): lane l holds row (l&15); K-chunks [kb..kb+7] and
// [kb+16..kb+23] with kb = (l>>4)*8  -> two contiguous 16B loads.
__device__ __forceinline__ v16h load_a_frag(const f16* p0, const f16* p1) {
  v8h lo = *(const v8h*)p0;
  v8h hi = *(const v8h*)p1;
  v16h r;
#pragma unroll
  for (int i = 0; i < 8; ++i) { r[i] = lo[i]; r[8 + i] = hi[i]; }
  return r;
}

// Wave-local LDS write->read fence (CDNA5 split counters: DScnt).
__device__ __forceinline__ void lds_sync_wave() {
  __builtin_amdgcn_wave_barrier();
  asm volatile("s_wait_dscnt 0" ::: "memory");
  __builtin_amdgcn_wave_barrier();
}

// ---------------------------------------------------------------- sort stage
__global__ void zero_counts_kernel(int* counts) { counts[threadIdx.x] = 0; }

__global__ void hist_kernel(const int* __restrict__ mask, int* __restrict__ counts) {
  int i = blockIdx.x * 256 + threadIdx.x;
  atomicAdd(&counts[mask[i] & (NGROUPS - 1)], 1);
}

__global__ void scan_kernel(const int* __restrict__ counts, int* __restrict__ poff) {
  if (threadIdx.x == 0) {
    int acc = 0;
    for (int g = 0; g < NGROUPS; ++g) {
      poff[g] = acc;
      int c = counts[g];
      if (c > NGCAP) c = NGCAP;
      acc += (c + 31) & ~31;
    }
    poff[NGROUPS] = acc;
  }
}

__global__ void fill_sidx_kernel(int* sidx) {
  sidx[blockIdx.x * 256 + threadIdx.x] = 0;
}

// Deterministic stable compaction: one block per group, in-order ballot scan.
__global__ __launch_bounds__(256) void compact_kernel(const int* __restrict__ mask,
                                                      const int* __restrict__ counts,
                                                      const int* __restrict__ poff,
                                                      int* __restrict__ sidx) {
  int g = blockIdx.x;
  __shared__ int wsum[8];
  __shared__ int sbase;
  int lane = threadIdx.x & 31, wave = threadIdx.x >> 5;
  if (threadIdx.x == 0) sbase = 0;
  __syncthreads();
  int base = poff[g];
  int cap = counts[g]; if (cap > NGCAP) cap = NGCAP;
  for (int start = 0; start < N_PTS; start += 256) {
    int i = start + (int)threadIdx.x;
    bool match = ((mask[i] & (NGROUPS - 1)) == g);
    unsigned long long bal = __ballot(match);
    int prefix = __popcll(bal & ((1ull << lane) - 1ull));
    int tot = __popcll(bal);
    if (lane == 0) wsum[wave] = tot;
    __syncthreads();
    int wbase = 0, btot = 0;
    for (int w = 0; w < 8; ++w) { if (w < wave) wbase += wsum[w]; btot += wsum[w]; }
    int rank = sbase + wbase + prefix;
    if (match && rank < cap) sidx[base + rank] = i;
    __syncthreads();
    if (threadIdx.x == 0) sbase += btot;
    __syncthreads();
  }
}

// ------------------------------------------------------------- prep kernels
__global__ void cvt_feat_kernel(const float* __restrict__ src, f16* __restrict__ dst) {
  size_t t = (size_t)blockIdx.x * 256 + threadIdx.x; // one float4 per thread
  float4 v = ((const float4*)src)[t];
  v4h o = {(f16)v.x, (f16)v.y, (f16)v.z, (f16)v.w};
  ((v4h*)dst)[t] = o;
}

// Pack a row-major [K x C] matrix into WMMA B-fragment layout:
// dst[((kt*CT + ct)*32 + lane)*16 + j] = src[(kt*32 + (lane>>4)*8 + ((j&8)<<1) + (j&7))*C + ct*16 + (lane&15)]
__global__ void packB_f32_kernel(const float* __restrict__ src, f16* __restrict__ dst, int C) {
  int tid = blockIdx.x * 256 + threadIdx.x;       // enumerates dst linearly
  int j = tid & 15;
  int lane = (tid >> 4) & 31;
  int rest = tid >> 9;                            // kt*CT + ct
  int CT = C >> 4;
  int ct = rest % CT, kt = rest / CT;
  int nn = lane & 15, kB = (lane >> 4) << 3;
  dst[tid] = (f16)src[(size_t)(kt * 32 + kB + ((j & 8) << 1) + (j & 7)) * C + ct * 16 + nn];
}

__global__ void packB_f16_kernel(const f16* __restrict__ src, f16* __restrict__ dst, int C) {
  int tid = blockIdx.x * 256 + threadIdx.x;
  int j = tid & 15;
  int lane = (tid >> 4) & 31;
  int rest = tid >> 9;
  int CT = C >> 4;
  int ct = rest % CT, kt = rest / CT;
  int nn = lane & 15, kB = (lane >> 4) << 3;
  dst[tid] = (f16)src[(size_t)(kt * 32 + kB + ((j & 8) << 1) + (j & 7)) * C + ct * 16 + nn];
}

// ------------------------------------------------------------- projections
// out[r, :] = feat16[sidx[r], :] @ W + bias  (f16 in / f32 accumulate / f16 out)
// One wave per 16x16 output tile, k-loop over DIM in steps of 32.
__global__ __launch_bounds__(32) void proj_kernel(const f16* __restrict__ feat,
                                                  const int* __restrict__ sidx,
                                                  const f16* __restrict__ wpack,
                                                  const float* __restrict__ bias,
                                                  f16* __restrict__ out, int cols) {
  int rt = blockIdx.x, ct = blockIdx.y;
  int lane = threadIdx.x;
  int m = lane & 15, half = lane >> 4;
  int row = sidx[rt * 16 + m];
  const f16* arow = feat + (size_t)row * DIM + half * 8;
  int CT = cols >> 4;
  size_t bstride = (size_t)CT * 512;              // halves per k-tile
  const f16* bp = wpack + ((size_t)ct * 32 + lane) * 16;
  v8f c = {};
#pragma unroll 4
  for (int kt = 0; kt < DIM / 32; ++kt) {
    v16h a = load_a_frag(arow + kt * 32, arow + kt * 32 + 16);
    v16h b = *(const v16h*)(bp + (size_t)kt * bstride);
    c = wmma_f16(a, b, c);
  }
  float bv = bias[ct * 16 + m];                   // C/D frag: lane -> col (lane&15)
#pragma unroll
  for (int v = 0; v < 8; ++v) {
    int r = v + 8 * half;                         // row within tile
    out[(size_t)(rt * 16 + r) * cols + ct * 16 + m] = (f16)(c[v] + bv);
  }
}

// --------------------------------------------------------------- attention
// grid (NGROUPS, 4), block 128 (4 waves). Each wave owns a 16-query tile.
__global__ __launch_bounds__(128) void attn_kernel(const f16* __restrict__ Qs,
                                                   const f16* __restrict__ Ks,
                                                   const f16* __restrict__ VsPack,
                                                   const int* __restrict__ counts,
                                                   const int* __restrict__ poff,
                                                   const int* __restrict__ sidx,
                                                   const float* __restrict__ features,
                                                   float* __restrict__ out) {
  __shared__ __align__(16) f16 sW[4][16][NGCAP];  // raw scores, then weights (57 KB)
  int g = blockIdx.x;
  int wave = threadIdx.x >> 5, lane = threadIdx.x & 31;
  int m = lane & 15, half = lane >> 4;
  int n = counts[g]; if (n > NGCAP) n = NGCAP;
  if (n == 0) return;
  int base = poff[g];
  int pn = (n + 31) & ~31;
  int tiles = pn >> 4;
  f16 (*myW)[NGCAP] = sW[wave];

  for (int rt = blockIdx.y * 4 + wave; rt < tiles; rt += 16) {
    // Q A-fragments for this 16-row tile (A=64 -> two k-steps of 32)
    const f16* qrow = Qs + (size_t)(base + rt * 16 + m) * ADIM + half * 8;
    v16h aq0 = load_a_frag(qrow, qrow + 16);
    v16h aq1 = load_a_frag(qrow + 32, qrow + 48);

    // ---- scores S = (Q K^T) * 1/sqrt(64); K^T B-frag = contiguous K rows
    for (int ct = 0; ct < tiles; ++ct) {
      const f16* krow = Ks + (size_t)(base + ct * 16 + m) * ADIM + half * 8;
      v16h b0 = load_a_frag(krow, krow + 16);
      v16h b1 = load_a_frag(krow + 32, krow + 48);
      v8f s = {};
      s = wmma_f16(aq0, b0, s);
      s = wmma_f16(aq1, b1, s);
      int col = ct * 16 + m;
      bool valid = col < n;                       // mask padded keys
#pragma unroll
      for (int v = 0; v < 8; ++v) {
        float sv = valid ? s[v] * 0.125f : -3.0e4f;
        myW[v + 8 * half][col] = (f16)sv;
      }
    }
    lds_sync_wave();

    // ---- softmax per row: lanes (l, l+16) split columns of row (l&15)
    float mx = -1.0e30f;
    for (int c = half; c < pn; c += 2) mx = fmaxf(mx, (float)myW[m][c]);
    mx = fmaxf(mx, __shfl_xor(mx, 16));
    float sum = 0.0f;
    for (int c = half; c < pn; c += 2) sum += __expf((float)myW[m][c] - mx);
    sum += __shfl_xor(sum, 16);
    float inv = 1.0f / sum;
    for (int c = half; c < pn; c += 2)
      myW[m][c] = (f16)(__expf((float)myW[m][c] - mx) * inv);
    lds_sync_wave();

    // ---- agg = weights @ V  (V pre-packed in B-fragment layout)
    int kts = pn >> 5;
    for (int ot = 0; ot < DIM / 16; ++ot) {
      v8f c8 = {};
      for (int kt = 0; kt < kts; ++kt) {
        const f16* wr = &myW[m][kt * 32 + half * 8];
        v16h aw = load_a_frag(wr, wr + 16);
        const f16* bp = VsPack + ((size_t)(((base >> 5) + kt) * (DIM / 16) + ot) * 32 + lane) * 16;
        v16h b = *(const v16h*)bp;
        c8 = wmma_f16(aw, b, c8);
      }
#pragma unroll
      for (int v = 0; v < 8; ++v) {
        int r = rt * 16 + v + 8 * half;
        if (r < n) {
          int orig = sidx[base + r];
          float val = c8[v];
          if (n == 1)  // singleton group keeps original features
            val = features[(size_t)orig * DIM + ot * 16 + m];
          out[(size_t)orig * DIM + ot * 16 + m] = val;
        }
      }
    }
  }
}

// ---------------------------------------------------------------- launcher
extern "C" void kernel_launch(void* const* d_in, const int* in_sizes, int n_in,
                              void* d_out, int out_size, void* d_ws, size_t ws_size,
                              hipStream_t stream) {
  const float* features = (const float*)d_in[0];
  const int*   mask     = (const int*)d_in[1];   // mask_ids (values 0..63)
  const float* Wq       = (const float*)d_in[2];
  const float* bq       = (const float*)d_in[3];
  const float* Wk       = (const float*)d_in[4];
  const float* bk       = (const float*)d_in[5];
  const float* Wv       = (const float*)d_in[6];
  const float* bv       = (const float*)d_in[7];
  float* out = (float*)d_out;

  char* w = (char*)d_ws;
  int* counts = (int*)w;            w += 256;
  int* poff   = (int*)w;            w += 512;
  int* sidx   = (int*)w;            w += (size_t)NRMAX * 4;
  f16* feat16 = (f16*)w;            w += (size_t)N_PTS * DIM * 2;
  f16* wqP    = (f16*)w;            w += (size_t)DIM * ADIM * 2;
  f16* wkP    = (f16*)w;            w += (size_t)DIM * ADIM * 2;
  f16* wvP    = (f16*)w;            w += (size_t)DIM * DIM * 2;
  f16* Qs     = (f16*)w;            w += (size_t)NRMAX * ADIM * 2;
  f16* Ks     = (f16*)w;            w += (size_t)NRMAX * ADIM * 2;
  f16* Vs     = (f16*)w;            w += (size_t)NRMAX * DIM * 2;
  f16* VsPack = (f16*)w;            w += (size_t)NRMAX * DIM * 2;  // ~31.2 MB total

  zero_counts_kernel<<<1, NGROUPS, 0, stream>>>(counts);
  hist_kernel<<<N_PTS / 256, 256, 0, stream>>>(mask, counts);
  scan_kernel<<<1, 1, 0, stream>>>(counts, poff);
  fill_sidx_kernel<<<NRMAX / 256, 256, 0, stream>>>(sidx);
  compact_kernel<<<NGROUPS, 256, 0, stream>>>(mask, counts, poff, sidx);

  cvt_feat_kernel<<<(N_PTS * DIM / 4) / 256, 256, 0, stream>>>(features, feat16);
  packB_f32_kernel<<<(DIM * ADIM) / 256, 256, 0, stream>>>(Wq, wqP, ADIM);
  packB_f32_kernel<<<(DIM * ADIM) / 256, 256, 0, stream>>>(Wk, wkP, ADIM);
  packB_f32_kernel<<<(DIM * DIM) / 256, 256, 0, stream>>>(Wv, wvP, DIM);

  proj_kernel<<<dim3(NRMAX / 16, ADIM / 16), 32, 0, stream>>>(feat16, sidx, wqP, bq, Qs, ADIM);
  proj_kernel<<<dim3(NRMAX / 16, ADIM / 16), 32, 0, stream>>>(feat16, sidx, wkP, bk, Ks, ADIM);
  proj_kernel<<<dim3(NRMAX / 16, DIM / 16), 32, 0, stream>>>(feat16, sidx, wvP, bv, Vs, DIM);

  packB_f16_kernel<<<((size_t)NRMAX * DIM) / 256, 256, 0, stream>>>(Vs, VsPack, DIM);

  attn_kernel<<<dim3(NGROUPS, 4), 128, 0, stream>>>(Qs, Ks, VsPack, counts, poff, sidx,
                                                    features, out);
}
